// saAxialBlock_53326313947744
// MI455X (gfx1250) — matrix operand order repaired and checked by reading, not compiled, not run on hardware
//
#include <hip/hip_runtime.h>

// Problem dims (B,C,D,H,W) = (2,16,128,80,80), fp32 in/out.
#define BB 2
#define CC 16
#define DD 128
#define HH 80
#define WW 80
#define HW 6400              // H*W
#define DHW (DD * HW)        // 819200
#define CDHW (CC * DHW)      // 13107200
#define FTOT (CC * HW)       // 102400
#define NCHUNK 200           // F-chunks per batch for QK^T
#define KC (FTOT / NCHUNK)   // 512 floats per chunk
#define KSTEPS (KC / 32)     // 16 WMMA k-steps per chunk
#define PITCH 40             // LDS row pitch in bf16 (80B) -> conflict-friendly
#define NFT (FTOT / 128)     // 800 f-tiles per batch for PV

typedef __attribute__((ext_vector_type(16))) __bf16 v16bf;
typedef __attribute__((ext_vector_type(8)))  __bf16 v8bf;
typedef __attribute__((ext_vector_type(8)))  float  v8f;

#if __has_builtin(__builtin_amdgcn_sched_barrier)
#define SCHED_FENCE() __builtin_amdgcn_sched_barrier(0)
#else
#define SCHED_FENCE()
#endif

// Native bf16 converts: clang selects v_cvt_pk_bf16_f32 (verified in disasm).
__device__ __forceinline__ unsigned pk2(float a, float b) {
  union { __bf16 h[2]; unsigned u; } r;
  r.h[0] = (__bf16)a;
  r.h[1] = (__bf16)b;
  return r.u;
}

__device__ __forceinline__ v16bf cat8(v8bf lo, v8bf hi) {
  return __builtin_shufflevector(lo, hi, 0,1,2,3,4,5,6,7,8,9,10,11,12,13,14,15);
}

// ---------------------------------------------------------------------------
// Kernel 1: partial S = Q(chunk) * K(chunk)^T  -> spart[b][chunk][128][128]
// grid = B*NCHUNK blocks, 256 threads (8 waves). Wave w owns output rows
// 16w..16w+15 (8 accumulators across the 128 columns).
// ---------------------------------------------------------------------------
__global__ __launch_bounds__(256)
void qk_partial_kernel(const float* __restrict__ x_,
                       const float* __restrict__ x_t,
                       float* __restrict__ spart) {
  const int blk   = blockIdx.x;
  const int b     = blk / NCHUNK;
  const int chunk = blk - b * NCHUNK;
  const float* Qb = x_  + (size_t)b * CDHW;
  const float* Kb = x_t + (size_t)b * CDHW;

  __shared__ __attribute__((aligned(16))) unsigned short ktile[128 * PITCH];

  const int tid  = threadIdx.x;
  const int wave = tid >> 5;
  const int lane = tid & 31;
  const int half = lane >> 4;
  const int l15  = lane & 15;

  v8f acc[8];
#pragma unroll
  for (int i = 0; i < 8; ++i)
#pragma unroll
    for (int j = 0; j < 8; ++j) acc[i][j] = 0.0f;

  for (int ks = 0; ks < KSTEPS; ++ks) {
    const int f  = chunk * KC + ks * 32;          // 32-aligned: never crosses c
    const int c  = f / HW;
    const int hw = f - c * HW;
    const float* qbase = Qb + (size_t)c * DHW + hw;
    const float* kbase = Kb + (size_t)c * DHW + hw;

    // --- stage K tile (128 rows x 32 f) -> LDS bf16 [row][k], pitch 40 ---
    {
      const int q4 = tid & 3;        // 4 lanes per row -> 64B/row/instr
      const int r0 = tid >> 2;       // 0..63
#pragma unroll
      for (int rb = 0; rb < 128; rb += 64) {
        const int row = rb + r0;
#pragma unroll
        for (int pass = 0; pass < 2; ++pass) {
          const float4 vv = *reinterpret_cast<const float4*>(
              kbase + (size_t)row * HW + q4 * 4 + pass * 16);
          uint2 pk;
          pk.x = pk2(vv.x, vv.y);
          pk.y = pk2(vv.z, vv.w);
          *reinterpret_cast<uint2*>(&ktile[row * PITCH + q4 * 4 + pass * 16]) = pk;
        }
      }
    }
    __syncthreads();

    // --- A fragment (Q rows) straight from global, half-interleaved layout:
    //     elem i -> k = (i&8)*2 + half*8 + (i&7)
    const float* qp = qbase + (size_t)(wave * 16 + l15) * HW + half * 8;
    const float4 a0 = *reinterpret_cast<const float4*>(qp);
    const float4 a1 = *reinterpret_cast<const float4*>(qp + 4);
    const float4 a2 = *reinterpret_cast<const float4*>(qp + 16);
    const float4 a3 = *reinterpret_cast<const float4*>(qp + 20);
    v16bf af;
    af[0]  = (__bf16)a0.x; af[1]  = (__bf16)a0.y;
    af[2]  = (__bf16)a0.z; af[3]  = (__bf16)a0.w;
    af[4]  = (__bf16)a1.x; af[5]  = (__bf16)a1.y;
    af[6]  = (__bf16)a1.z; af[7]  = (__bf16)a1.w;
    af[8]  = (__bf16)a2.x; af[9]  = (__bf16)a2.y;
    af[10] = (__bf16)a2.z; af[11] = (__bf16)a2.w;
    af[12] = (__bf16)a3.x; af[13] = (__bf16)a3.y;
    af[14] = (__bf16)a3.z; af[15] = (__bf16)a3.w;

    // --- load ALL 8 B fragments (kept in distinct regs by the sched fence),
    //     then run the 8 WMMAs so LDS latency overlaps the matrix pipe.
    v16bf bf[8];
#pragma unroll
    for (int nt = 0; nt < 8; ++nt) {
      const unsigned short* bp = &ktile[(nt * 16 + l15) * PITCH + half * 8];
      bf[nt] = cat8(*reinterpret_cast<const v8bf*>(bp),
                    *reinterpret_cast<const v8bf*>(bp + 16));
    }
    SCHED_FENCE();   // keep the ds_load clause ahead of the WMMA chain
#pragma unroll
    for (int nt = 0; nt < 8; ++nt) {
      acc[nt] = __builtin_amdgcn_wmma_f32_16x16x32_bf16(
          false, af, false, bf[nt], (short)0, acc[nt], false, false);
    }
    __syncthreads();
  }

  // store partial 128x128 tile (C/D layout: VGPR v -> M = v + 8*half, N = l15)
  float* sp = spart + (size_t)(b * NCHUNK + chunk) * (128 * 128);
#pragma unroll
  for (int nt = 0; nt < 8; ++nt)
#pragma unroll
    for (int v = 0; v < 8; ++v) {
      const int m = wave * 16 + v + 8 * half;
      sp[m * 128 + nt * 16 + l15] = acc[nt][v];
    }
}

// ---------------------------------------------------------------------------
// Kernel 2: reduce NCHUNK partials + row softmax -> P as bf16
// grid = B*128 blocks (one row each), 128 threads.
// ---------------------------------------------------------------------------
__global__ __launch_bounds__(128)
void softmax_kernel(const float* __restrict__ spart,
                    unsigned short* __restrict__ pbuf) {
  const int b = blockIdx.x >> 7;
  const int d = blockIdx.x & 127;
  const int t = threadIdx.x;

  const float* base = spart + (size_t)b * NCHUNK * (128 * 128) + d * 128 + t;
  float s = 0.0f;
  for (int ch = 0; ch < NCHUNK; ++ch) s += base[(size_t)ch * (128 * 128)];

  __shared__ float red[128];
  red[t] = s;
  __syncthreads();
#pragma unroll
  for (int off = 64; off > 0; off >>= 1) {
    if (t < off) red[t] = fmaxf(red[t], red[t + off]);
    __syncthreads();
  }
  const float m = red[0];
  __syncthreads();
  const float e = __expf(s - m);
  red[t] = e;
  __syncthreads();
#pragma unroll
  for (int off = 64; off > 0; off >>= 1) {
    if (t < off) red[t] += red[t + off];
    __syncthreads();
  }
  const float inv = 1.0f / red[0];
  union { __bf16 h; unsigned short u; } r;
  r.h = (__bf16)(e * inv);
  pbuf[(size_t)b * (128 * 128) + d * 128 + t] = r.u;
}

// ---------------------------------------------------------------------------
// Kernel 3: O = P * V, f-tiles of 128. grid = B*NFT blocks, 256 threads.
// V k-tiles staged to LDS transposed ([f][e]); P fragments from global (L2).
// ---------------------------------------------------------------------------
__global__ __launch_bounds__(256)
void pv_kernel(const float* __restrict__ g_x,
               const unsigned short* __restrict__ pbuf,
               float* __restrict__ out) {
  const int blk = blockIdx.x;
  const int b   = blk / NFT;
  const int ft  = blk - b * NFT;
  const int f0  = ft * 128;                 // 6400 = 50*128: never crosses c
  const int c   = f0 / HW;
  const int hw  = f0 - c * HW;
  const float* Vb = g_x + (size_t)b * CDHW + (size_t)c * DHW + hw;
  float*       Ob = out + (size_t)b * CDHW + (size_t)c * DHW + hw;
  const unsigned short* Pb = pbuf + (size_t)b * (128 * 128);

  __shared__ __attribute__((aligned(16))) unsigned short vtile[128 * PITCH];

  const int tid  = threadIdx.x;
  const int wave = tid >> 5;
  const int lane = tid & 31;
  const int half = lane >> 4;
  const int l15  = lane & 15;

  v8f acc[8];
#pragma unroll
  for (int i = 0; i < 8; ++i)
#pragma unroll
    for (int j = 0; j < 8; ++j) acc[i][j] = 0.0f;

  for (int ks = 0; ks < 4; ++ks) {
    const int e0 = ks * 32;
    // --- stage V (32 e x 128 f) -> LDS transposed [f][e], coalesced 512B reads
    {
      const int fl = tid & 127;
      const int eh = (tid >> 7) * 16;
#pragma unroll
      for (int i = 0; i < 16; i += 2) {
        const float x0 = Vb[(size_t)(e0 + eh + i)     * HW + fl];
        const float x1 = Vb[(size_t)(e0 + eh + i + 1) * HW + fl];
        *reinterpret_cast<unsigned*>(&vtile[fl * PITCH + eh + i]) = pk2(x0, x1);
      }
    }
    __syncthreads();

    // --- A fragment from P (global, interleaved A layout, 16B aligned)
    const unsigned short* pr = Pb + (wave * 16 + l15) * 128 + e0 + half * 8;
    v8bf alo = *reinterpret_cast<const v8bf*>(pr);
    v8bf ahi = *reinterpret_cast<const v8bf*>(pr + 16);
    const v16bf af = cat8(alo, ahi);

    // --- load all 8 B fragments, fence, then 8 WMMAs
    v16bf bf[8];
#pragma unroll
    for (int nt = 0; nt < 8; ++nt) {
      const unsigned short* bp = &vtile[(nt * 16 + l15) * PITCH + half * 8];
      bf[nt] = cat8(*reinterpret_cast<const v8bf*>(bp),
                    *reinterpret_cast<const v8bf*>(bp + 16));
    }
    SCHED_FENCE();
#pragma unroll
    for (int nt = 0; nt < 8; ++nt) {
      acc[nt] = __builtin_amdgcn_wmma_f32_16x16x32_bf16(
          false, af, false, bf[nt], (short)0, acc[nt], false, false);
    }
    __syncthreads();
  }

  // --- store O in [B,C,D,H,W] order: 2x64B segments per store instruction
#pragma unroll
  for (int nt = 0; nt < 8; ++nt)
#pragma unroll
    for (int v = 0; v < 8; ++v) {
      const int dd = wave * 16 + v + 8 * half;
      Ob[(size_t)dd * HW + nt * 16 + l15] = acc[nt][v];
    }
}

// ---------------------------------------------------------------------------
extern "C" void kernel_launch(void* const* d_in, const int* in_sizes, int n_in,
                              void* d_out, int out_size, void* d_ws, size_t ws_size,
                              hipStream_t stream) {
  const float* x_  = (const float*)d_in[0];
  const float* x_t = (const float*)d_in[1];
  const float* g_x = (const float*)d_in[2];
  float* out = (float*)d_out;

  // d_ws layout: [ spart : B*NCHUNK*128*128 fp32 (~25.6MB) | pbuf : B*128*128 bf16 ]
  float* spart = (float*)d_ws;
  unsigned short* pbuf = (unsigned short*)((char*)d_ws +
      (size_t)BB * NCHUNK * 128 * 128 * sizeof(float));

  qk_partial_kernel<<<dim3(BB * NCHUNK), dim3(256), 0, stream>>>(x_, x_t, spart);
  softmax_kernel<<<dim3(BB * 128), dim3(128), 0, stream>>>(spart, pbuf);
  pv_kernel<<<dim3(BB * NFT), dim3(256), 0, stream>>>(g_x, pbuf, out);
}
